// PARAFORMER_DECODER_83837761618112
// MI455X (gfx1250) — matrix operand order repaired and checked by reading, not compiled, not run on hardware
//
#include <hip/hip_runtime.h>

// ---------------------------------------------------------------------------
// Paraformer-large streaming decoder for MI455X (gfx1250, wave32, WMMA).
// All GEMMs: fp32 operands streamed from HBM, converted to bf16 while staging
// into LDS, computed with v_wmma_f32_16x16x32_bf16 (fp32 accumulate).
// ---------------------------------------------------------------------------

typedef __attribute__((ext_vector_type(16))) __bf16 bf16x16;
typedef __attribute__((ext_vector_type(8)))  float  f32x8;

static constexpr int L_   = 16;
static constexpr int H_   = 512;
static constexpr int DFF_ = 2048;
static constexpr int NH_  = 4;
static constexpr int DK_  = 128;
static constexpr int K_   = 11;
static constexpr int V_   = 8404;
static constexpr int T_   = 64;
static constexpr int S_   = 2048;
static constexpr int CKV_ = 256;
static constexpr int LB_  = 256;
static constexpr int CT_  = CKV_ + S_;   // 2304

#define BM 64
#define BN 128
#define BK 32
#define LDSP 40   // 32 + 8 bf16 pad -> 80B row pitch, 16B-aligned frag chunks

union FragBF {
  bf16x16 v;
  uint4   q[2];
  __bf16  e[16];
};

// ---------------------------------------------------------------------------
// Generic batched GEMM: C = epilogue(A[M,K] @ B[K,N]), bf16 WMMA, f32 accum.
// Tile 64x128, 8 waves, each wave computes a 32x32 block (2x2 frags, 4 WMMA
// per K-step). epilogue: (+bias[col]) * scale, optional relu, residual add.
// ---------------------------------------------------------------------------
__global__ __launch_bounds__(256)
void gemm_wmma_bf16(const float* __restrict__ A, const float* __restrict__ B,
                    const float* __restrict__ bias, const float* __restrict__ resid,
                    float* __restrict__ C,
                    int M, int N, int Kd, int lda, int ldb, int ldc,
                    long sA, long sB, long sC, float scale, int relu) {
  __shared__ __align__(16) __bf16 As[BM * LDSP];
  __shared__ __align__(16) __bf16 Bs[BN * LDSP];   // N-major (transposed) tile

  const int tid  = threadIdx.x;
  const int lane = tid & 31;
  const int wid  = tid >> 5;

  const float* Ab = A + (long)blockIdx.z * sA + (long)blockIdx.y * BM * lda;
  const float* Bb = B + (long)blockIdx.z * sB;
  float*       Cb = C + (long)blockIdx.z * sC;
  const float* Rb = resid ? (resid + (long)blockIdx.z * sC) : nullptr;
  const int n0blk = blockIdx.x * BN;

  // wave -> 32x32 output block: 2 M-frag rows x 4 N-wave cols
  const int wm = (wid & 1) * 32;        // wave row offset in tile (0/32)
  const int wn = (wid >> 1) * 32;       // wave col offset in tile (0..96)

  f32x8 acc00 = {}, acc01 = {}, acc10 = {}, acc11 = {};

  // staging assignments (256 threads)
  const int arow = tid >> 2;            // 0..63 : A row in tile
  const int akk  = (tid & 3) * 8;       // 0,8,16,24 : A k-chunk
  const int bn   = tid & 127;           // 0..127 : B column in tile
  const int bkk  = (tid >> 7) * 16;     // 0 or 16 : B k-chunk

  // B column handling (branchless OOB): clamp pointer, zero via mask
  const int  bcol  = n0blk + bn;
  const int  bcolc = bcol < N ? bcol : (N - 1);
  const float bmask = bcol < N ? 1.0f : 0.0f;

  // fragment lane decomposition
  const int fr  = lane & 15;
  const int fhA = (lane >> 4) * 8;      // A: K chunks {fhA, 16+fhA}
  const int fhB = (lane >> 4) * 8;      // B: K chunks {fhB, 16+fhB}

  const __bf16* pa0 = &As[(wm + fr) * LDSP];
  const __bf16* pa1 = &As[(wm + 16 + fr) * LDSP];
  const __bf16* pb0 = &Bs[(wn + fr) * LDSP + fhB];
  const __bf16* pb1 = &Bs[(wn + 16 + fr) * LDSP + fhB];

  for (int k0 = 0; k0 < Kd; k0 += BK) {
    __syncthreads();
    { // ---- stage A tile (coalesced float4 loads, cvt to bf16) ----
      const float* src = Ab + (long)arow * lda + k0 + akk;
      const float4 f0 = ((const float4*)src)[0];
      const float4 f1 = ((const float4*)src)[1];
      FragBF t;
      t.e[0] = (__bf16)f0.x; t.e[1] = (__bf16)f0.y;
      t.e[2] = (__bf16)f0.z; t.e[3] = (__bf16)f0.w;
      t.e[4] = (__bf16)f1.x; t.e[5] = (__bf16)f1.y;
      t.e[6] = (__bf16)f1.z; t.e[7] = (__bf16)f1.w;
      *(uint4*)&As[arow * LDSP + akk] = t.q[0];
      __builtin_prefetch(src + BK, 0, 3);            // next K tile, same row
    }
    { // ---- stage B tile: one column, 16 K values, branchless ----
      const float* src = Bb + (long)(k0 + bkk) * ldb + bcolc;
      FragBF t;
#pragma unroll
      for (int j = 0; j < 16; ++j)
        t.e[j] = (__bf16)(src[(long)j * ldb] * bmask);
      uint4* dst = (uint4*)&Bs[bn * LDSP + bkk];
      dst[0] = t.q[0];
      dst[1] = t.q[1];
      __builtin_prefetch(src + (long)BK * ldb, 0, 3);
    }
    __syncthreads();

    // ---- fragment loads (ISA VGPR layouts) + 4 WMMAs ----
    FragBF a0, a1, b0, b1;
    a0.q[0] = *(const uint4*)(pa0 + fhA);            // K = hi*8 .. +7
    a0.q[1] = *(const uint4*)(pa0 + 16 + fhA);       // K = 16+hi*8 .. +7
    a1.q[0] = *(const uint4*)(pa1 + fhA);
    a1.q[1] = *(const uint4*)(pa1 + 16 + fhA);
    b0.q[0] = *(const uint4*)(pb0);
    b0.q[1] = *(const uint4*)(pb0 + 16);
    b1.q[0] = *(const uint4*)(pb1);
    b1.q[1] = *(const uint4*)(pb1 + 16);

    acc00 = __builtin_amdgcn_wmma_f32_16x16x32_bf16(false, a0.v, false, b0.v,
                                                    (short)0, acc00, false, false);
    acc01 = __builtin_amdgcn_wmma_f32_16x16x32_bf16(false, a0.v, false, b1.v,
                                                    (short)0, acc01, false, false);
    acc10 = __builtin_amdgcn_wmma_f32_16x16x32_bf16(false, a1.v, false, b0.v,
                                                    (short)0, acc10, false, false);
    acc11 = __builtin_amdgcn_wmma_f32_16x16x32_bf16(false, a1.v, false, b1.v,
                                                    (short)0, acc11, false, false);
  }

  // ---- epilogue: C/D layout -> lanes 0-15: N=lane,M=r ; 16-31: M=8+r ----
  const int mb = (lane >> 4) * 8;
  f32x8 accs[2][2] = {{acc00, acc01}, {acc10, acc11}};
#pragma unroll
  for (int fm = 0; fm < 2; ++fm) {
#pragma unroll
    for (int fn = 0; fn < 2; ++fn) {
      const int col = n0blk + wn + fn * 16 + (lane & 15);
      if (col < N) {
        const float bv = bias ? bias[col] : 0.0f;
#pragma unroll
        for (int i = 0; i < 8; ++i) {
          const int m = blockIdx.y * BM + wm + fm * 16 + mb + i;
          float v = (accs[fm][fn][i] + bv) * scale;
          if (relu) v = fmaxf(v, 0.0f);
          if (Rb) v += Rb[(long)m * ldc + col];
          Cb[(long)m * ldc + col] = v;
        }
      }
    }
  }
}

// ---------------------------------------------------------------------------
// LayerNorm over last dim (block per row, wave32 shuffle + LDS reduce).
// Safe in-place (out may alias in).
// ---------------------------------------------------------------------------
__global__ __launch_bounds__(256)
void ln_kernel(const float* in, float* out,
               const float* __restrict__ g, const float* __restrict__ b, int C) {
  const float* x = in  + (long)blockIdx.x * C;
  float*       y = out + (long)blockIdx.x * C;
  float s = 0.f, s2 = 0.f;
  for (int i = threadIdx.x; i < C; i += blockDim.x) {
    const float v = x[i];
    s += v; s2 += v * v;
  }
#pragma unroll
  for (int off = 16; off > 0; off >>= 1) {
    s  += __shfl_down(s, off, 32);
    s2 += __shfl_down(s2, off, 32);
  }
  __shared__ float rs[8], rs2[8];
  const int wid = threadIdx.x >> 5, lane = threadIdx.x & 31;
  if (lane == 0) { rs[wid] = s; rs2[wid] = s2; }
  __syncthreads();
  if (threadIdx.x == 0) {
    float ts = 0.f, ts2 = 0.f;
    const int nw = blockDim.x >> 5;
    for (int w = 0; w < nw; ++w) { ts += rs[w]; ts2 += rs2[w]; }
    const float mean = ts / C;
    rs[0]  = mean;
    rs2[0] = rsqrtf(fmaxf(ts2 / C - mean * mean, 0.f) + 1e-5f);
  }
  __syncthreads();
  const float mean = rs[0], rstd = rs2[0];
  for (int i = threadIdx.x; i < C; i += blockDim.x)
    y[i] = (x[i] - mean) * rstd * g[i] + b[i];
}

// ---------------------------------------------------------------------------
// Row softmax, in place.
// ---------------------------------------------------------------------------
__global__ __launch_bounds__(256)
void softmax_kernel(float* x_, int C) {
  float* x = x_ + (long)blockIdx.x * C;
  __shared__ float red[8];
  __shared__ float bval;
  const int wid = threadIdx.x >> 5, lane = threadIdx.x & 31;
  const int nw = blockDim.x >> 5;

  float m = -3.4e38f;
  for (int i = threadIdx.x; i < C; i += blockDim.x) m = fmaxf(m, x[i]);
#pragma unroll
  for (int off = 16; off > 0; off >>= 1) m = fmaxf(m, __shfl_down(m, off, 32));
  if (lane == 0) red[wid] = m;
  __syncthreads();
  if (threadIdx.x == 0) {
    float t = red[0];
    for (int w = 1; w < nw; ++w) t = fmaxf(t, red[w]);
    bval = t;
  }
  __syncthreads();
  m = bval;

  float s = 0.f;
  for (int i = threadIdx.x; i < C; i += blockDim.x) {
    const float e = __expf(x[i] - m);
    x[i] = e;
    s += e;
  }
#pragma unroll
  for (int off = 16; off > 0; off >>= 1) s += __shfl_down(s, off, 32);
  __syncthreads();
  if (lane == 0) red[wid] = s;
  __syncthreads();
  if (threadIdx.x == 0) {
    float t = 0.f;
    for (int w = 0; w < nw; ++w) t += red[w];
    bval = 1.f / t;
  }
  __syncthreads();
  const float inv = bval;
  for (int i = threadIdx.x; i < C; i += blockDim.x) x[i] *= inv;
}

// ---------------------------------------------------------------------------
// FSMN depthwise conv (K=11) over [cache(K-1) | y^T(T)] per channel h,
// x = conv^T + y + residual ; also emits save_fsmn (= x_cat row).
// One block per channel h.
// ---------------------------------------------------------------------------
__global__ __launch_bounds__(128)
void fsmn_kernel(const float* __restrict__ cache, const float* __restrict__ y,
                 const float* __restrict__ resid, const float* __restrict__ w,
                 float* __restrict__ x, float* __restrict__ save) {
  const int h = blockIdx.x;
  __shared__ float xc[K_ - 1 + T_];   // 74
  __shared__ float wv[K_];
  for (int j = threadIdx.x; j < K_ - 1 + T_; j += blockDim.x)
    xc[j] = (j < K_ - 1) ? cache[h * (K_ - 1) + j] : y[(j - (K_ - 1)) * H_ + h];
  if (threadIdx.x < K_) wv[threadIdx.x] = w[h * K_ + threadIdx.x];
  __syncthreads();
  for (int j = threadIdx.x; j < K_ - 1 + T_; j += blockDim.x)
    save[h * (K_ - 1 + T_) + j] = xc[j];
  if (threadIdx.x < T_) {
    const int t = threadIdx.x;
    float acc = 0.f;
#pragma unroll
    for (int i = 0; i < K_; ++i) acc += xc[t + i] * wv[i];
    x[t * H_ + h] = acc + y[t * H_ + h] + resid[t * H_ + h];
  }
}

// ---------------------------------------------------------------------------
// k_full[h][d][c] = c<CKV ? kcache : kv[s, h*DK+d] ; last LB cols -> save_keys
// ---------------------------------------------------------------------------
__global__ __launch_bounds__(256)
void assemble_k_kernel(const float* __restrict__ kv, const float* __restrict__ kc,
                       float* __restrict__ kfull, float* __restrict__ skeys) {
  const int idx = blockIdx.x * blockDim.x + threadIdx.x;
  if (idx >= NH_ * DK_ * CT_) return;
  const int c  = idx % CT_;
  const int hd = idx / CT_;                // h*DK + d
  float v;
  if (c < CKV_) {
    v = kc[hd * CKV_ + c];
  } else {
    const int s = c - CKV_;
    const int h = hd / DK_, d = hd % DK_;
    v = kv[(long)s * (2 * H_) + h * DK_ + d];
  }
  kfull[idx] = v;
  if (c >= CT_ - LB_) skeys[hd * LB_ + (c - (CT_ - LB_))] = v;
}

// ---------------------------------------------------------------------------
// v_full[h][r][d] = r<CKV ? vcache : kv[s, H + h*DK+d] ; last LB rows -> save
// ---------------------------------------------------------------------------
__global__ __launch_bounds__(256)
void assemble_v_kernel(const float* __restrict__ kv, const float* __restrict__ vc,
                       float* __restrict__ vfull, float* __restrict__ svals) {
  const int idx = blockIdx.x * blockDim.x + threadIdx.x;
  if (idx >= NH_ * CT_ * DK_) return;
  const int d = idx % DK_;
  const int r = (idx / DK_) % CT_;
  const int h = idx / (DK_ * CT_);
  float v;
  if (r < CKV_) v = vc[(h * CKV_ + r) * DK_ + d];
  else          v = kv[(long)(r - CKV_) * (2 * H_) + H_ + h * DK_ + d];
  vfull[idx] = v;
  if (r >= CT_ - LB_) svals[(h * LB_ + (r - (CT_ - LB_))) * DK_ + d] = v;
}

// ---------------------------------------------------------------------------
// Row argmax (first index wins on ties, matching jnp.argmax).
// ---------------------------------------------------------------------------
__global__ __launch_bounds__(256)
void argmax_kernel(const float* __restrict__ logits, int* __restrict__ ids, int C) {
  const float* x = logits + (long)blockIdx.x * C;
  __shared__ float sm[256];
  __shared__ int   si[256];
  float bm = -3.4e38f;
  int   bi = 0x7fffffff;
  for (int i = threadIdx.x; i < C; i += blockDim.x) {
    const float v = x[i];
    if (v > bm) { bm = v; bi = i; }
  }
  sm[threadIdx.x] = bm; si[threadIdx.x] = bi;
  __syncthreads();
  for (int off = 128; off > 0; off >>= 1) {
    if ((int)threadIdx.x < off) {
      const float vo = sm[threadIdx.x + off];
      const int   io = si[threadIdx.x + off];
      if (vo > sm[threadIdx.x] ||
          (vo == sm[threadIdx.x] && io < si[threadIdx.x])) {
        sm[threadIdx.x] = vo; si[threadIdx.x] = io;
      }
    }
    __syncthreads();
  }
  if (threadIdx.x == 0) ids[blockIdx.x] = si[0];
}

// ---------------------------------------------------------------------------
// Host orchestration
// ---------------------------------------------------------------------------
extern "C" void kernel_launch(void* const* d_in, const int* in_sizes, int n_in,
                              void* d_out, int out_size, void* d_ws, size_t ws_size,
                              hipStream_t stream) {
  (void)in_sizes; (void)n_in; (void)out_size; (void)ws_size;

  const float* enc         = (const float*)d_in[0];
  const float* list_frame  = (const float*)d_in[1];
  const float* fsmn_cache  = (const float*)d_in[2];
  const float* key_cache   = (const float*)d_in[3];
  const float* value_cache = (const float*)d_in[4];
  const float* norm1_g = (const float*)d_in[5];
  const float* norm1_b = (const float*)d_in[6];
  const float* norm2_g = (const float*)d_in[7];
  const float* norm2_b = (const float*)d_in[8];
  const float* norm3_g = (const float*)d_in[9];
  const float* norm3_b = (const float*)d_in[10];
  const float* ffn_norm_g = (const float*)d_in[11];
  const float* ffn_norm_b = (const float*)d_in[12];
  const float* w1 = (const float*)d_in[13];
  const float* b1 = (const float*)d_in[14];
  const float* w2 = (const float*)d_in[15];
  const float* fsmn_w = (const float*)d_in[16];
  const float* qw = (const float*)d_in[17];
  const float* qbv = (const float*)d_in[18];
  const float* kvw = (const float*)d_in[19];
  const float* kvb = (const float*)d_in[20];
  const float* ow = (const float*)d_in[21];
  const float* ob = (const float*)d_in[22];
  const float* d3_norm1_g = (const float*)d_in[23];
  const float* d3_norm1_b = (const float*)d_in[24];
  const float* d3_w1 = (const float*)d_in[25];
  const float* d3_b1 = (const float*)d_in[26];
  const float* d3_ffn_norm_g = (const float*)d_in[27];
  const float* d3_ffn_norm_b = (const float*)d_in[28];
  const float* d3_w2 = (const float*)d_in[29];
  const float* after_norm_g = (const float*)d_in[30];
  const float* after_norm_b = (const float*)d_in[31];
  const float* out_w = (const float*)d_in[32];
  const float* out_b = (const float*)d_in[33];

  // outputs (flat concat: save_fsmn, save_keys, save_values, ids)
  float* outF      = (float*)d_out;
  float* save_fsmn = outF;
  float* save_keys = save_fsmn + (size_t)L_ * H_ * (K_ - 1 + T_);
  float* save_vals = save_keys + (size_t)L_ * NH_ * DK_ * LB_;
  int*   ids       = (int*)(save_vals + (size_t)L_ * NH_ * LB_ * DK_);

  // workspace layout (~23.6 MB of fp32)
  float* ws    = (float*)d_ws;
  float* lf    = ws; ws += T_ * H_;
  float* yb    = ws; ws += T_ * H_;
  float* xb    = ws; ws += T_ * H_;
  float* xln   = ws; ws += T_ * H_;
  float* qbuf  = ws; ws += T_ * H_;
  float* ctx   = ws; ws += T_ * H_;
  float* t1    = ws; ws += (size_t)T_ * DFF_;
  float* kv    = ws; ws += (size_t)S_ * 2 * H_;
  float* kfull = ws; ws += (size_t)NH_ * DK_ * CT_;
  float* vfull = ws; ws += (size_t)NH_ * CT_ * DK_;
  float* sc    = ws; ws += (size_t)NH_ * T_ * CT_;
  float* lg    = ws; ws += (size_t)T_ * V_;

  hipMemcpyAsync(lf, list_frame, sizeof(float) * T_ * H_,
                 hipMemcpyDeviceToDevice, stream);

  auto gemm = [&](const float* A, const float* B, const float* bias,
                  const float* resid, float* C, int M, int N, int Kd,
                  int lda, int ldb, int ldc, long sA, long sB, long sC,
                  int batch, float scale, int relu) {
    dim3 grid((N + BN - 1) / BN, M / BM, batch);
    gemm_wmma_bf16<<<grid, dim3(256), 0, stream>>>(
        A, B, bias, resid, C, M, N, Kd, lda, ldb, ldc, sA, sB, sC, scale, relu);
  };
  auto ln = [&](const float* in, float* out, const float* g, const float* b,
                int rows, int C) {
    ln_kernel<<<rows, 256, 0, stream>>>(in, out, g, b, C);
  };

  const float qscale = 0.08838834764831845f;  // DK^-0.5

  for (int l = 0; l < L_; ++l) {
    // ---- FFN block: y = ln2( ln_ffn(relu(ln1(lf) @ W1 + B1)) @ W2 ) ----
    ln(lf, xln, norm1_g + l * H_, norm1_b + l * H_, T_, H_);
    gemm(xln, w1 + (size_t)l * H_ * DFF_, b1 + (size_t)l * DFF_, nullptr, t1,
         T_, DFF_, H_, H_, DFF_, DFF_, 0, 0, 0, 1, 1.f, 1);
    ln(t1, t1, ffn_norm_g + (size_t)l * DFF_, ffn_norm_b + (size_t)l * DFF_, T_, DFF_);
    gemm(t1, w2 + (size_t)l * DFF_ * H_, nullptr, nullptr, yb,
         T_, H_, DFF_, DFF_, H_, H_, 0, 0, 0, 1, 1.f, 0);
    ln(yb, yb, norm2_g + l * H_, norm2_b + l * H_, T_, H_);

    // ---- FSMN memory block: x = conv(x_cat)^T + y + lf ----
    fsmn_kernel<<<H_, 128, 0, stream>>>(
        fsmn_cache + (size_t)l * H_ * (K_ - 1), yb, lf,
        fsmn_w + (size_t)l * H_ * K_, xb,
        save_fsmn + (size_t)l * H_ * (K_ - 1 + T_));

    // ---- cross attention ----
    ln(xb, xln, norm3_g + l * H_, norm3_b + l * H_, T_, H_);
    gemm(xln, qw + (size_t)l * H_ * H_, qbv + (size_t)l * H_, nullptr, qbuf,
         T_, H_, H_, H_, H_, H_, 0, 0, 0, 1, qscale, 0);
    // kv = enc @ KVW + KVB   (dominant GEMM: 2048x1024x512)
    gemm(enc, kvw + (size_t)l * H_ * 2 * H_, kvb + (size_t)l * 2 * H_, nullptr, kv,
         S_, 2 * H_, H_, H_, 2 * H_, 2 * H_, 0, 0, 0, 1, 1.f, 0);
    assemble_k_kernel<<<(NH_ * DK_ * CT_ + 255) / 256, 256, 0, stream>>>(
        kv, key_cache + (size_t)l * NH_ * DK_ * CKV_, kfull,
        save_keys + (size_t)l * NH_ * DK_ * LB_);
    assemble_v_kernel<<<(NH_ * CT_ * DK_ + 255) / 256, 256, 0, stream>>>(
        kv, value_cache + (size_t)l * NH_ * CKV_ * DK_, vfull,
        save_vals + (size_t)l * NH_ * LB_ * DK_);
    // scores[h] = q_h @ k_full_h   (batched over heads)
    gemm(qbuf, kfull, nullptr, nullptr, sc,
         T_, CT_, DK_, H_, CT_, CT_,
         (long)DK_, (long)DK_ * CT_, (long)T_ * CT_, NH_, 1.f, 0);
    softmax_kernel<<<NH_ * T_, 256, 0, stream>>>(sc, CT_);
    // ctx[h] = attn_h @ v_full_h
    gemm(sc, vfull, nullptr, nullptr, ctx,
         T_, DK_, CT_, CT_, DK_, H_,
         (long)T_ * CT_, (long)CT_ * DK_, (long)DK_, NH_, 1.f, 0);
    // lf = x + ctx @ OW + OB
    gemm(ctx, ow + (size_t)l * H_ * H_, ob + (size_t)l * H_, xb, lf,
         T_, H_, H_, H_, H_, H_, 0, 0, 0, 1, 1.f, 0);
  }

  // ---- decoder3 + output head ----
  ln(lf, xln, d3_norm1_g, d3_norm1_b, T_, H_);
  gemm(xln, d3_w1, d3_b1, nullptr, t1, T_, DFF_, H_, H_, DFF_, DFF_, 0, 0, 0, 1, 1.f, 1);
  ln(t1, t1, d3_ffn_norm_g, d3_ffn_norm_b, T_, DFF_);
  gemm(t1, d3_w2, nullptr, nullptr, yb, T_, H_, DFF_, DFF_, H_, H_, 0, 0, 0, 1, 1.f, 0);
  ln(yb, xln, after_norm_g, after_norm_b, T_, H_);
  gemm(xln, out_w, out_b, nullptr, lg, T_, V_, H_, H_, V_, V_, 0, 0, 0, 1, 1.f, 0);
  argmax_kernel<<<T_, 256, 0, stream>>>(lg, ids, V_);
}
